// demo_net_26834955665436
// MI455X (gfx1250) — matrix operand order repaired and checked by reading, not compiled, not run on hardware
//
#include <hip/hip_runtime.h>
#include <hip/hip_bf16.h>
#include <math.h>
#include <stdint.h>

// Problem constants (from reference)
#define TT 192   // visits
#define DD 128   // med_dim
#define HH 512   // hidden
#define VV 1957  // voc size

typedef __attribute__((ext_vector_type(16))) _Float16 v16h;
typedef __attribute__((ext_vector_type(8)))  _Float16 v8h;
typedef __attribute__((ext_vector_type(8)))  float    v8f;

#if __has_builtin(__builtin_amdgcn_tanhf)
#define FAST_TANH(x) __builtin_amdgcn_tanhf(x)
#else
#define FAST_TANH(x) tanhf(x)
#endif

#if __has_builtin(__builtin_amdgcn_s_wait_asynccnt)
#define WAIT_ASYNC(n) __builtin_amdgcn_s_wait_asynccnt(n)
#else
#define WAIT_ASYNC(n) asm volatile("s_wait_asynccnt %0" ::"i"(n) : "memory")
#endif

// gfx1250 async copy: global memory -> LDS, 16B per lane, tracked by ASYNCcnt.
__device__ __forceinline__ void async_copy_b128(unsigned lds_off, uint64_t gaddr) {
    asm volatile("global_load_async_to_lds_b128 %0, %1, off"
                 :: "v"(lds_off), "v"(gaddr) : "memory");
}

// Triangular tile base: tiles (i, jt) for jt <= i>>4, 16KB each.
__device__ __host__ __forceinline__ int tile_base(int i) {
    int g = i >> 4, r = i & 15;
    return i + 8 * g * (g - 1) + g * r;
}
#define NTILES 1248  // sum over i of (i/16 + 1)

// ---------------------------------------------------------------------------
// Kernel 1: afold[i,h] = pr[i,:] @ W1[:D, h] + b1[h]
//           bmem[j,h]  = his[j,:] @ W1[D:, h]  (his[0]=0, his[j]=pr[j-1])
// ---------------------------------------------------------------------------
__global__ __launch_bounds__(256) void prep_ab(const float* __restrict__ pr,
                                               const float* __restrict__ W1,
                                               const float* __restrict__ b1,
                                               float* __restrict__ afold,
                                               float* __restrict__ bmem) {
    int idx   = blockIdx.x * 256 + threadIdx.x;   // exact: 768 blocks
    int which = idx / (TT * HH);
    int r     = idx % (TT * HH);
    int i = r / HH, h = r % HH;
    float s = 0.f;
    if (which == 0) {
        for (int k = 0; k < DD; ++k) s += pr[i * DD + k] * W1[k * HH + h];
        afold[i * HH + h] = s + b1[h];
    } else {
        if (i == 0) {
            bmem[h] = 0.f;
        } else {
            for (int k = 0; k < DD; ++k) s += pr[(i - 1) * DD + k] * W1[(DD + k) * HH + h];
            bmem[i * HH + h] = s;
        }
    }
}

// ---------------------------------------------------------------------------
// Kernel 2: W2 -> f16, B-fragment swizzled, zero-padded to 2048 cols.
// Layout [nt(128)][kt(16)][lane(32)][e(16)]:
//   n = nt*16 + lane%16 ; k = kt*32 + 8*(lane/16) + (e<8 ? e : 8+e)
// ---------------------------------------------------------------------------
__global__ __launch_bounds__(256) void swz_w2(const float* __restrict__ W2,
                                              _Float16* __restrict__ w2s) {
    int idx  = blockIdx.x * 256 + threadIdx.x;    // exact: 4096 blocks
    int e    = idx & 15;
    int lane = (idx >> 4) & 31;
    int kt   = (idx >> 9) & 15;
    int nt   = idx >> 13;
    int n    = nt * 16 + (lane & 15);
    int k    = kt * 32 + ((lane >> 4) << 3) + ((e < 8) ? e : (8 + e));
    float v  = (n < VV) ? W2[k * VV + n] : 0.f;
    w2s[idx] = (_Float16)v;
}

// ---------------------------------------------------------------------------
// Kernel 3: precompute h tiles ONCE (kills the 16x redundant tanh work).
// Tile (i, jt): 16 rows j = jt*16..+15, 512 K, f16, WMMA A-fragment swizzle
// [kt(16)][lane(32)][e(16)]:
//   row m = lane%16 ; kb = (lane<16 ? 0 : 8)
//   k = kt*32 + (e<8 ? kb+e : 16+kb+(e-8))
// ---------------------------------------------------------------------------
__global__ __launch_bounds__(256) void h_build(const float* __restrict__ afold,
                                               const float* __restrict__ bmem,
                                               _Float16* __restrict__ hswz) {
    __shared__ float sA[HH];
    const int i = blockIdx.y, jt = blockIdx.x;
    const int g = i >> 4;
    if (jt > g) return;                            // uniform: whole block exits
    const int tile = tile_base(i) + jt;
    const int jb = jt * 16;
    for (int k = threadIdx.x; k < HH; k += 256) sA[k] = afold[i * HH + k];
    __syncthreads();
    _Float16* dst = hswz + (size_t)tile * 8192;
    for (int t = threadIdx.x; t < 8192; t += 256) {
        int e = t & 15, lane = (t >> 4) & 31, kt = t >> 9;
        int jl = lane & 15;
        int kb = (lane & 16) ? 8 : 0;
        int kl = (e < 8) ? (kb + e) : (16 + kb + (e - 8));
        int k  = kt * 32 + kl;
        float x = sA[k] + bmem[(jb + jl) * HH + k];
        dst[t] = (_Float16)FAST_TANH(x);
    }
}

// ---------------------------------------------------------------------------
// Kernel 4: main GEMM. grid=(16 vgroups, 192 i), 8 waves; wave owns 16 cols.
// - B fragments manually hoisted to 128 VGPRs (asm "memory" clobbers would
//   otherwise defeat LICM and force per-iteration reloads).
// - h tile double-buffered into LDS via global_load_async_to_lds_b128.
// - drug stream loaded non-temporally (single-use, keep L2 for w2s/hswz).
// ---------------------------------------------------------------------------
__global__ __launch_bounds__(256) void gate_main_pre(const _Float16* __restrict__ hswz,
                                                     const _Float16* __restrict__ w2s,
                                                     const float* __restrict__ b2,
                                                     const float* __restrict__ drug,
                                                     float* __restrict__ out) {
    __shared__ _Float16 sT[2][8192];               // 2 x 16KB double buffer

    const int i    = blockIdx.y;
    const int tid  = threadIdx.x;
    const int wave = tid >> 5;
    const int lane = tid & 31;
    const int nt   = blockIdx.x * 8 + wave;
    const int n16  = lane & 15;
    const int hi16 = lane >> 4;
    const int v    = nt * 16 + n16;

    const int tb0   = tile_base(i);
    const int ntile = (i >> 4) + 1;

    const float b2v = (v < VV) ? b2[v] : 0.f;
    const int   vcl = (v < VV) ? v : (VV - 1);

    // Hoist all 16 B fragments into registers (loop-invariant; 128 VGPRs).
    v16h Bf[16];
    #pragma unroll
    for (int kt = 0; kt < 16; ++kt)
        Bf[kt] = *(const v16h*)(w2s + (((size_t)(nt * 16 + kt)) * 32 + lane) * 16);

    // Each wave async-copies 2KB of the 16KB tile (4 x 512B wave-instrs).
    const unsigned my_off = (unsigned)(wave * 2048 + lane * 16);
    const unsigned lds0   = (unsigned)(size_t)&sT[0][0];
    const unsigned lds1   = (unsigned)(size_t)&sT[1][0];

    auto issue = [&](int buf, int jt) {
        const char* gsrc = (const char*)(hswz + (size_t)(tb0 + jt) * 8192);
        unsigned lbase = (buf ? lds1 : lds0) + my_off;
        #pragma unroll
        for (int q = 0; q < 4; ++q)
            async_copy_b128(lbase + q * 512, (uint64_t)(uintptr_t)(gsrc + my_off + q * 512));
    };

    float acc = 0.f;
    issue(0, 0);

    for (int jt = 0; jt < ntile; ++jt) {
        const int jb = jt * 16;
        if (jt + 1 < ntile) {                      // uniform branch
            issue((jt + 1) & 1, jt + 1);
            WAIT_ASYNC(4);                         // current tile's 4 done
        } else {
            WAIT_ASYNC(0);
        }
        __syncthreads();                           // all waves' copies visible

        const _Float16* tbuf = &sT[jt & 1][0];
        v8f c = {};
        #pragma unroll
        for (int kt = 0; kt < 16; ++kt) {
            v16h a = *(const v16h*)(tbuf + (kt * 32 + lane) * 16);   // 2x ds_load_b128
            c = __builtin_amdgcn_wmma_f32_16x16x32_f16(
                    false, a, false, Bf[kt], (short)0, c, false, false);
        }

        float p = 0.f;
        const float* drow = drug + ((size_t)i * TT + jb) * VV + vcl;
        #pragma unroll
        for (int r = 0; r < 8; ++r) {
            int jl = r + 8 * hi16;                 // D-matrix row for this VGPR
            int j  = jb + jl;
            float s = FAST_TANH(c[r] + b2v);
            float d = (j <= i && v < VV)
                        ? __builtin_nontemporal_load(drow + (size_t)jl * VV) : 0.f;
            p += s * d;
        }
        p += __shfl_xor(p, 16, 32);
        acc += p;
        __syncthreads();                           // reads done before buf reuse
    }

    if (lane < 16 && v < VV)
        __builtin_nontemporal_store(acc, out + (size_t)i * VV + v);
}

// ---------------------------------------------------------------------------
// Fallback (fused) kernel, used when ws_size can't hold the h precompute.
// ---------------------------------------------------------------------------
__global__ __launch_bounds__(256) void gate_fused(const float* __restrict__ afold,
                                                  const float* __restrict__ bmem,
                                                  const _Float16* __restrict__ w2s,
                                                  const float* __restrict__ b2,
                                                  const float* __restrict__ drug,
                                                  float* __restrict__ out) {
    __shared__ float    sA[HH];
    __shared__ _Float16 sH[16][HH];

    const int i    = blockIdx.y;
    const int tid  = threadIdx.x;
    const int wave = tid >> 5;
    const int lane = tid & 31;
    const int nt   = blockIdx.x * 8 + wave;
    const int n16  = lane & 15;
    const int hi16 = lane >> 4;
    const int v    = nt * 16 + n16;

    for (int k = tid; k < HH; k += 256) sA[k] = afold[i * HH + k];

    const float b2v = (v < VV) ? b2[v] : 0.f;
    const int   vcl = (v < VV) ? v : (VV - 1);

    v16h Bf[16];
    #pragma unroll
    for (int kt = 0; kt < 16; ++kt)
        Bf[kt] = *(const v16h*)(w2s + (((size_t)(nt * 16 + kt)) * 32 + lane) * 16);

    float acc = 0.f;

    for (int jb = 0; jb <= i; jb += 16) {
        __syncthreads();
        for (int t = tid; t < 16 * HH; t += 256) {
            int jl = t >> 9;
            int k  = t & (HH - 1);
            float x = sA[k] + bmem[(jb + jl) * HH + k];
            sH[jl][k] = (_Float16)FAST_TANH(x);
        }
        __syncthreads();

        v8f c = {};
        const int m  = n16;
        const int kb = hi16 ? 8 : 0;
        #pragma unroll
        for (int kt = 0; kt < 16; ++kt) {
            const _Float16* rp = &sH[m][kt * 32 + kb];
            v8h lo = *(const v8h*)rp;
            v8h hh = *(const v8h*)(rp + 16);
            v16h a;
            #pragma unroll
            for (int e = 0; e < 8; ++e) { a[e] = lo[e]; a[8 + e] = hh[e]; }
            c = __builtin_amdgcn_wmma_f32_16x16x32_f16(
                    false, a, false, Bf[kt], (short)0, c, false, false);
        }

        float p = 0.f;
        const float* drow = drug + ((size_t)i * TT + jb) * VV + vcl;
        #pragma unroll
        for (int r = 0; r < 8; ++r) {
            int jl = r + 8 * hi16;
            int j  = jb + jl;
            float s = FAST_TANH(c[r] + b2v);
            float d = (j <= i && v < VV)
                        ? __builtin_nontemporal_load(drow + (size_t)jl * VV) : 0.f;
            p += s * d;
        }
        p += __shfl_xor(p, 16, 32);
        acc += p;
    }

    if (lane < 16 && v < VV)
        __builtin_nontemporal_store(acc, out + (size_t)i * VV + v);
}

// ---------------------------------------------------------------------------
extern "C" void kernel_launch(void* const* d_in, const int* in_sizes, int n_in,
                              void* d_out, int out_size, void* d_ws, size_t ws_size,
                              hipStream_t stream) {
    const float* pr   = (const float*)d_in[0];   // [T, D]
    const float* drug = (const float*)d_in[1];   // [T, T, V]
    const float* W1   = (const float*)d_in[2];   // [2D, H]
    const float* b1   = (const float*)d_in[3];   // [H]
    const float* W2   = (const float*)d_in[4];   // [H, V]
    const float* b2   = (const float*)d_in[5];   // [V]
    float* out = (float*)d_out;                  // [T, V]

    // Workspace: afold | bmem | w2s (f16 swizzled) | hswz (f16 tiles)
    float*    afold = (float*)d_ws;
    float*    bmem  = afold + TT * HH;
    _Float16* w2s   = (_Float16*)(bmem + TT * HH);
    _Float16* hswz  = (_Float16*)((char*)d_ws + 2883584);  // after the above

    const size_t need = 2883584ull + (size_t)NTILES * 16384ull;  // ~23.3 MB

    prep_ab<<<(2 * TT * HH) / 256, 256, 0, stream>>>(pr, W1, b1, afold, bmem);
    swz_w2<<<(128 * 16 * 32 * 16) / 256, 256, 0, stream>>>(W2, w2s);

    if (ws_size >= need) {
        dim3 gh(12, TT);
        h_build<<<gh, 256, 0, stream>>>(afold, bmem, hswz);
        dim3 gm(16, TT);
        gate_main_pre<<<gm, 256, 0, stream>>>(hswz, w2s, b2, drug, out);
    } else {
        dim3 gm(16, TT);
        gate_fused<<<gm, 256, 0, stream>>>(afold, bmem, w2s, b2, drug, out);
    }
}
